// SRLSecondOrderScorer_44659069943937
// MI455X (gfx1250) — compile-verified, matrix operand order checked by reading, not run on hardware
//
#include <hip/hip_runtime.h>

// ---------------------------------------------------------------------------
// SRL second-order scorer for MI455X (gfx1250, wave32, WMMA bf16 16x16x32)
// B=2, S=192, D_IN=1024, D=128
// All GEMM operands staged in LDS row-major per fragment row (m- or n-major),
// pitch-padded to 136 bf16 so every fragment is 2x ds_load_b128, ~conflict-free.
// ---------------------------------------------------------------------------

#define SEQ 192
#define DIM 128
#define DIN 1024
#define BS  384        // B*SEQ rows
#define LDP 136        // LDS row pitch in bf16 elements (272B, 16B aligned)

typedef __attribute__((ext_vector_type(16))) __bf16 v16bf;
typedef __attribute__((ext_vector_type(8)))  float  v8f;

static __device__ __forceinline__ v8f wmma_bf16(v16bf a, v16bf b, v8f c) {
  // D = A(16x32 bf16) * B(32x16 bf16) + C(16x16 f32)
  return __builtin_amdgcn_wmma_f32_16x16x32_bf16(
      /*neg_a=*/false, a, /*neg_b=*/false, b,
      /*c_mod=*/(short)0, c, /*reuse_a=*/false, /*reuse_b=*/false);
}

// Fragment loader for A (16x32, rows=m) or B (32x16, rows=n) from a row-major
// bf16 buffer with row pitch ldp. ISA 16-bit layout (wave32): lane row=lane&15,
// lane bit4 selects the 8-wide K-half; per lane the 16 values are two
// contiguous 8-element K runs -> two 16B LDS loads, no packing VALU.
static __device__ __forceinline__ v16bf load_frag(const __bf16* base, int ldp, int lane) {
  int m  = lane & 15;
  int hi = (lane >> 4) & 1;
  const __bf16* p = base + m * ldp + (hi << 3);
  union { uint4 q[2]; v16bf v; } u;
  u.q[0] = *(const uint4*)(p);        // k = hi*8 .. hi*8+7      (frag 0..7)
  u.q[1] = *(const uint4*)(p + 16);   // k = 16+hi*8 .. +7       (frag 8..15)
  return u.v;
}

// Copy nrows rows of 128 contiguous bf16 into pitch-LDP LDS rows (16B chunks).
static __device__ __forceinline__ void copy_rows_pitched(
    __bf16* dst, const __bf16* src, int nrows, int tid, int nth) {
  int total = nrows * 16;  // 16 uint4 per 128-elem row
  for (int i = tid; i < total; i += nth) {
    int r = i >> 4, c = (i & 15) << 3;
    *(uint4*)(dst + r * LDP + c) = *(const uint4*)(src + r * DIM + c);
  }
}

// ---------------------------------------------------------------------------
// Kernel 1: p/sh/st = leaky_relu(x @ W + b, 0.1), stored bf16.
// grid (384 rows, 3 heads), 128 threads.
// ---------------------------------------------------------------------------
__global__ __launch_bounds__(128) void mlp_bf16_kernel(
    const float* __restrict__ x,
    const float* __restrict__ W0, const float* __restrict__ b0,
    const float* __restrict__ W1, const float* __restrict__ b1,
    const float* __restrict__ W2, const float* __restrict__ b2,
    __bf16* __restrict__ actbf) {  // [3][384][128]
  __shared__ float xrow[DIN];
  int row = blockIdx.x;
  int sel = blockIdx.y;
  const float* W    = (sel == 0) ? W0 : (sel == 1) ? W1 : W2;
  const float* bias = (sel == 0) ? b0 : (sel == 1) ? b1 : b2;
  int tid = threadIdx.x;

  const float4* xs = (const float4*)(x + (size_t)row * DIN);
  float4*       xd = (float4*)xrow;
  for (int i = tid; i < DIN / 4; i += 128) xd[i] = xs[i];
  __syncthreads();

  float acc = bias[tid];
#pragma unroll 8
  for (int k = 0; k < DIN; ++k) acc += xrow[k] * W[(size_t)k * DIM + tid];
  float v = acc > 0.0f ? acc : 0.1f * acc;
  actbf[((size_t)sel * BS + row) * DIM + tid] = (__bf16)v;
}

// ---------------------------------------------------------------------------
// Kernel 2: wzT[t,bz,j,i] = sum_k Z[bz,k] * W_t[i,k,j]   (bf16 WMMA)
// grid (i=128, t=6), 256 threads (8 waves); wave: 3 M-tiles x 8 N-tiles.
// W slice is transposed to [j][k] at LDS fill so B fragments are b128 loads;
// output stored j-major so kernel 3's LDS fill is a straight pitched copy.
// ---------------------------------------------------------------------------
__global__ __launch_bounds__(256) void wz_gemm_kernel(
    const __bf16* __restrict__ actbf,     // [3][384][128]
    const float* __restrict__ Wq0, const float* __restrict__ Wq1,
    const float* __restrict__ Wq2, const float* __restrict__ Wq3,
    const float* __restrict__ Wq4, const float* __restrict__ Wq5,
    __bf16* __restrict__ wzbf) {          // [6][384][128(j)][128(i)]
  __shared__ __bf16 Zs[BS * LDP];         // 102 KB  rows = bz
  __shared__ __bf16 WsT[DIM * LDP];       //  34 KB  rows = j, cols = k

  int i = blockIdx.x;   // 0..127  (slice of W over index i)
  int t = blockIdx.y;   // 0..5    (which triaffine head)
  const float* Wsel = (t == 0) ? Wq0 : (t == 1) ? Wq1 : (t == 2) ? Wq2
                    : (t == 3) ? Wq3 : (t == 4) ? Wq4 : Wq5;
  int zsel = (t < 4) ? 0 : (t == 4 ? 1 : 2);  // z-arg: p,p,p,p,sh,st
  int tid = threadIdx.x;

  copy_rows_pitched(Zs, actbf + (size_t)zsel * BS * DIM, BS, tid, 256);
  const float* wsrc = Wsel + (size_t)i * DIM * DIM;  // contiguous [k][j] slice
  for (int idx = tid; idx < DIM * DIM; idx += 256) {
    int k = idx >> 7, j = idx & 127;                 // coalesced global read
    WsT[j * LDP + k] = (__bf16)wsrc[idx];            // transpose into LDS
  }
  __syncthreads();

  int wv = tid >> 5, lane = tid & 31;
#pragma unroll 1
  for (int mt = wv * 3; mt < wv * 3 + 3; ++mt) {
    int mbase = mt * 16;
#pragma unroll 1
    for (int nt = 0; nt < 8; ++nt) {
      int nbase = nt * 16;
      v8f acc = {};
#pragma unroll
      for (int kt = 0; kt < 4; ++kt) {
        v16bf a  = load_frag(Zs + mbase * LDP + kt * 32, LDP, lane);
        v16bf bm = load_frag(WsT + nbase * LDP + kt * 32, LDP, lane);
        acc = wmma_bf16(a, bm, acc);
      }
      int n = lane & 15, hi = (lane >> 4) & 1;
#pragma unroll
      for (int r = 0; r < 8; ++r) {
        int m = mbase + r + hi * 8;                  // bz row
        int j = nbase + n;
        wzbf[(((size_t)t * BS + m) * DIM + j) * DIM + i] = (__bf16)acc[r];
      }
    }
  }
}

// ---------------------------------------------------------------------------
// Kernel 3: per (z, b, head): S = (X @ wz) @ Y^T, symmetrize / permute.
// grid (z=192, b=2, o=6), 192 threads (6 waves).
//   o=0: sym(tri(sh,st,p,W_span_ph))   out [b,z,x,y]
//   o=1: sym(tri(sh,st,p,W_span_pt))   out [b,z,x,y]
//   o=2: sym(tri(sh,sh,p,W_ph_sib))    out [b,z,x,y]
//   o=3: sym(tri(st,st,p,W_pt_sib))    out [b,z,x,y]
//   o=4: tri(p,p,sh,W_ph_cop)          out [b,x,y,z]   (permuted)
//   o=5: sym(tri(p,p,st,W_pt_cop))     out [b,x,y,z]   (permuted)
// ---------------------------------------------------------------------------
__global__ __launch_bounds__(192) void triaffine_score_kernel(
    const __bf16* __restrict__ actbf,   // [3][384][128]
    const __bf16* __restrict__ wzbf,    // [6][384][128(j)][128(i)]
    float* __restrict__ out) {          // [6][2][192][192][192]
  __shared__ __bf16 Xs[SEQ * LDP];      // 51 KB  rows = x
  __shared__ __bf16 Ys[SEQ * LDP];      // 51 KB  rows = y
  __shared__ __bf16 WzsT[DIM * LDP];    // 34 KB  rows = j, cols = i
  __shared__ __bf16 Ts[SEQ * LDP];      // 51 KB  rows = x, cols = j
  __shared__ float  stage[6][256];      //  6 KB  (one 16x16 f32 tile per wave)

  int z = blockIdx.x, b = blockIdx.y, o = blockIdx.z;
  int xsel, ysel;
  switch (o) {
    case 0: case 1: xsel = 1; ysel = 2; break;  // sh, st
    case 2:         xsel = 1; ysel = 1; break;  // sh, sh
    case 3:         xsel = 2; ysel = 2; break;  // st, st
    default:        xsel = 0; ysel = 0; break;  // p, p
  }
  const bool sym  = (o != 4);
  const bool perm = (o >= 4);

  int tid = threadIdx.x;
  copy_rows_pitched(Xs, actbf + ((size_t)xsel * BS + b * SEQ) * DIM, SEQ, tid, 192);
  const __bf16* Yp = Xs;
  if (ysel != xsel) {
    copy_rows_pitched(Ys, actbf + ((size_t)ysel * BS + b * SEQ) * DIM, SEQ, tid, 192);
    Yp = Ys;
  }
  copy_rows_pitched(WzsT, wzbf + ((size_t)o * BS + b * SEQ + z) * DIM * DIM,
                    DIM, tid, 192);
  __syncthreads();

  int wv = tid >> 5, lane = tid & 31;

  // ---- GEMM1: T[x][j] = sum_i X[x][i] * wz[i][j];  B(k=i,n=j) = WzsT[j][i] --
#pragma unroll 1
  for (int mt = wv * 2; mt < wv * 2 + 2; ++mt) {
    int mbase = mt * 16;
#pragma unroll 1
    for (int nt = 0; nt < 8; ++nt) {
      int nbase = nt * 16;
      v8f acc = {};
#pragma unroll
      for (int kt = 0; kt < 4; ++kt) {
        v16bf a  = load_frag(Xs + mbase * LDP + kt * 32, LDP, lane);
        v16bf bm = load_frag(WzsT + nbase * LDP + kt * 32, LDP, lane);
        acc = wmma_bf16(a, bm, acc);
      }
      int n = lane & 15, hi = (lane >> 4) & 1;
#pragma unroll
      for (int r = 0; r < 8; ++r)
        Ts[(mbase + r + hi * 8) * LDP + nbase + n] = (__bf16)acc[r];
    }
  }
  __syncthreads();

  // ---- GEMM2: S[x][y] = sum_j T[x][j] * Y[y][j];  B(k=j,n=y) = Y[y][j] -----
  const size_t S3    = (size_t)SEQ * SEQ * SEQ;
  float*       obase = out + ((size_t)o * 2 + b) * S3;
  const int    nt12  = SEQ / 16;                       // 12 tile rows/cols
  const int    ntiles = sym ? (nt12 * (nt12 + 1)) / 2  // upper tri incl diag
                            : nt12 * nt12;

#pragma unroll 1
  for (int idx = wv; idx < ntiles; idx += 6) {
    int xt, yt;
    if (sym) {
      int r = idx, xr = 0;
      while (r >= nt12 - xr) { r -= nt12 - xr; ++xr; }
      xt = xr; yt = xr + r;
    } else {
      xt = idx / nt12; yt = idx % nt12;
    }
    int xb = xt * 16, yb = yt * 16;

    v8f acc = {};
#pragma unroll
    for (int kt = 0; kt < 4; ++kt) {
      v16bf a  = load_frag(Ts + xb * LDP + kt * 32, LDP, lane);
      v16bf bm = load_frag(Yp + yb * LDP + kt * 32, LDP, lane);
      acc = wmma_bf16(a, bm, acc);
    }

    // Stage C tile (lane n=lane&15, VGPR r -> m=r+8*(lane>>4)) into LDS.
    int n = lane & 15, hi = (lane >> 4) & 1;
#pragma unroll
    for (int r = 0; r < 8; ++r) stage[wv][(r + hi * 8) * 16 + n] = acc[r];
    __builtin_amdgcn_wave_barrier();  // keep DS store->load order

#pragma unroll 1
    for (int e = lane; e < 256; e += 32) {
      int   rr = e >> 4, cc = e & 15;
      float val = stage[wv][e];
      int   xg = xb + rr, yg = yb + cc;
      if (!perm) {  // out[b][z][x][y]
        if (sym && xt == yt) {
          float v2 = (rr <= cc) ? val : stage[wv][cc * 16 + rr];
          obase[((size_t)z * SEQ + xg) * SEQ + yg] = v2;
        } else {
          obase[((size_t)z * SEQ + xg) * SEQ + yg] = val;
          if (sym) obase[((size_t)z * SEQ + yg) * SEQ + xg] = val;
        }
      } else {      // out[b][x][y][z]
        if (sym && xt == yt) {
          float v2 = (rr <= cc) ? val : stage[wv][cc * 16 + rr];
          obase[((size_t)xg * SEQ + yg) * SEQ + z] = v2;
        } else {
          obase[((size_t)xg * SEQ + yg) * SEQ + z] = val;
          if (sym) obase[((size_t)yg * SEQ + xg) * SEQ + z] = val;
        }
      }
    }
    __builtin_amdgcn_wave_barrier();
  }
}

// ---------------------------------------------------------------------------
// Launch
// ---------------------------------------------------------------------------
extern "C" void kernel_launch(void* const* d_in, const int* in_sizes, int n_in,
                              void* d_out, int out_size, void* d_ws, size_t ws_size,
                              hipStream_t stream) {
  (void)in_sizes; (void)n_in; (void)out_size;

  const float* x  = (const float*)d_in[0];
  const float* Wp = (const float*)d_in[1];  const float* bp = (const float*)d_in[2];
  const float* Wh = (const float*)d_in[3];  const float* bh = (const float*)d_in[4];
  const float* Wt = (const float*)d_in[5];  const float* bt = (const float*)d_in[6];
  const float* Wspan_ph = (const float*)d_in[7];
  const float* Wspan_pt = (const float*)d_in[8];
  const float* Wph_sib  = (const float*)d_in[9];
  const float* Wpt_sib  = (const float*)d_in[10];
  const float* Wph_cop  = (const float*)d_in[11];
  const float* Wpt_cop  = (const float*)d_in[12];

  // Workspace: actbf [3][384][128] bf16, then wzT [6][384][128][128] bf16.
  const size_t act_bytes = (size_t)3 * BS * DIM * 2;               // 294,912
  const size_t wz_bytes  = (size_t)6 * BS * DIM * DIM * 2;         // 75,497,472
  if (ws_size < act_bytes + wz_bytes) return;                      // need ~76 MB
  __bf16* actbf = (__bf16*)d_ws;
  __bf16* wzbf  = (__bf16*)((char*)d_ws + act_bytes);

  mlp_bf16_kernel<<<dim3(BS, 3), 128, 0, stream>>>(
      x, Wp, bp, Wh, bh, Wt, bt, actbf);

  wz_gemm_kernel<<<dim3(DIM, 6), 256, 0, stream>>>(
      actbf, Wspan_ph, Wspan_pt, Wph_sib, Wpt_sib, Wph_cop, Wpt_cop, wzbf);

  triaffine_score_kernel<<<dim3(SEQ, 2, 6), 192, 0, stream>>>(
      actbf, wzbf, (float*)d_out);
}
